// SelfAttention_81922206204654
// MI455X (gfx1250) — compile-verified, matrix operand order checked by reading, not run on hardware
//
#include <hip/hip_runtime.h>

// ---------------------------------------------------------------------------
// MI455X (gfx1250) fused multi-head self-attention, f16 WMMA / f32 accum.
//   B=4, T=1024, EMBED=512, H=8, head_dim=512 (reference uses full EMBED/head)
// Workspace layout (148 MB total):
//   Qh  : (B,H,T,512) f16 @ 0      (scaled by 512^-0.25)
//   Kh  : (B,H,T,512) f16 @ 32MB   (scaled by 512^-0.25)
//   Vt  : (B,H,512,T) f16 @ 64MB   (transposed -> WMMA B-operand b128 loads)
//   Oh  : (B,T,4096)  f16 @ 96MB
//   Xh  : (4096,512)  f16 @ 128MB  (x converted once)
//   Wqh/Wkh/Wvh : (4096,512) f16 @ 132/136/140MB
//   Woh : (512,4096)  f16 @ 144MB
// All GEMM inner loops are pure f16 global_load_b128 + v_wmma (no cvt chains).
// ---------------------------------------------------------------------------

typedef _Float16 v16h __attribute__((ext_vector_type(16)));
typedef _Float16 v8h  __attribute__((ext_vector_type(8)));
typedef _Float16 v4h  __attribute__((ext_vector_type(4)));
typedef float    v8f  __attribute__((ext_vector_type(8)));
typedef float    v4f  __attribute__((ext_vector_type(4)));

#define WMMA(a, b, c) \
  __builtin_amdgcn_wmma_f32_16x16x32_f16(false, (a), false, (b), (short)0, (c), false, false)

// A/B fragment element e (lane L, hs=L>>4): K = (e>>3)*16 + hs*8 + (e&7)
// -> two contiguous 8-half (16B) runs at K0 = hs*8 and K0+16.
static __device__ __forceinline__ v16h frag_f16(const _Float16* p0) {
  v8h lo = *(const v8h*)p0;
  v8h hi = *(const v8h*)(p0 + 16);
  v16h r;
#pragma unroll
  for (int i = 0; i < 8; ++i) { r[i] = lo[i]; r[8 + i] = hi[i]; }
  return r;
}

// ---------------------------------------------------------------------------
// Kernel 0: one-shot f32 -> f16 conversion of x and the four weights.
// All five tensors are 2,097,152 elements. grid = (2048, 5), block = 256.
// ---------------------------------------------------------------------------
__global__ __launch_bounds__(256) void cvt_f16_kernel(
    const float* __restrict__ sx, const float* __restrict__ sq,
    const float* __restrict__ sk, const float* __restrict__ sv,
    const float* __restrict__ so,
    _Float16* __restrict__ dx, _Float16* __restrict__ dq,
    _Float16* __restrict__ dk, _Float16* __restrict__ dv,
    _Float16* __restrict__ dw) {
  const float* src;
  _Float16* dst;
  switch (blockIdx.y) {
    case 0: src = sx; dst = dx; break;
    case 1: src = sq; dst = dq; break;
    case 2: src = sk; dst = dk; break;
    case 3: src = sv; dst = dv; break;
    default: src = so; dst = dw; break;
  }
  const size_t i = ((size_t)blockIdx.x * 256 + threadIdx.x) * 4;
  v4f v = *(const v4f*)(src + i);
  v4h h;
#pragma unroll
  for (int j = 0; j < 4; ++j) h[j] = (_Float16)v[j];
  *(v4h*)(dst + i) = h;
}

// ---------------------------------------------------------------------------
// Kernel 1: Q/K/V projections (f16 x f16 -> f32). grid = (64, 64, 3); block 128.
// Each wave: 16(M) x 64(N) strip, K over 512 in steps of 32 (16x4 WMMAs).
// ---------------------------------------------------------------------------
__global__ __launch_bounds__(128) void qkv_proj_kernel(
    const _Float16* __restrict__ Xh, const _Float16* __restrict__ Wqh,
    const _Float16* __restrict__ Wkh, const _Float16* __restrict__ Wvh,
    _Float16* __restrict__ Qh, _Float16* __restrict__ Kh,
    _Float16* __restrict__ Vt) {
  const int lane = threadIdx.x & 31, wave = threadIdx.x >> 5;
  const int lm = lane & 15, hs = lane >> 4;
  const int z = blockIdx.z;
  const _Float16* W = (z == 0) ? Wqh : ((z == 1) ? Wkh : Wvh);
  const int m0 = (blockIdx.y * 4 + wave) * 16;  // row in (b*t) space
  const int n0 = blockIdx.x * 64;               // col in (h*d) space

  v8f acc[4] = {};
  const _Float16* xrow = Xh + (size_t)(m0 + lm) * 512;
#pragma unroll 4
  for (int kk = 0; kk < 512; kk += 32) {
    v16h a = frag_f16(xrow + kk + hs * 8);
#pragma unroll
    for (int j = 0; j < 4; ++j) {
      // B[c][n] = W[n][c]: lane holds W row n, contiguous in c
      v16h bfr = frag_f16(W + (size_t)(n0 + j * 16 + lm) * 512 + kk + hs * 8);
      acc[j] = WMMA(a, bfr, acc[j]);
    }
  }
  const float sc = (z < 2) ? 0.21022410381342865f : 1.0f;  // 512^-0.25
#pragma unroll
  for (int j = 0; j < 4; ++j) {
    const int n = n0 + j * 16 + lm;
    const int h = n >> 9, d = n & 511;
#pragma unroll
    for (int r = 0; r < 8; ++r) {
      const int m = m0 + r + hs * 8;  // C layout: row = r + 8*hs, col = lm
      const int bb = m >> 10, t = m & 1023;
      const _Float16 v = (_Float16)(acc[j][r] * sc);
      if (z == 0)
        Qh[(((size_t)(bb * 8 + h)) * 1024 + t) * 512 + d] = v;
      else if (z == 1)
        Kh[(((size_t)(bb * 8 + h)) * 1024 + t) * 512 + d] = v;
      else
        Vt[(((size_t)(bb * 8 + h)) * 512 + d) * 1024 + t] = v;  // transposed
    }
  }
}

// ---------------------------------------------------------------------------
// Kernel 2: flash attention. grid = (T/16, B*H); block = 128 (4 waves).
// Waves split d=512 into 128-wide slices for the O accumulator; S partials
// are reduced across waves in LDS; 16 threads run the online softmax.
// ---------------------------------------------------------------------------
__global__ __launch_bounds__(128) void flash_attn_kernel(
    const _Float16* __restrict__ Qh, const _Float16* __restrict__ Kh,
    const _Float16* __restrict__ Vt, _Float16* __restrict__ Oh) {
  __shared__ float Spart[4 * 16 * 32];              // per-wave S partials
  __shared__ __align__(16) _Float16 Ptile[16 * 32]; // softmax probs (A frag src)
  __shared__ float rowmax[16], rowsum[16], alphaS[16];

  const int tid = threadIdx.x;
  const int lane = tid & 31, wave = tid >> 5;
  const int lm = lane & 15, hs = lane >> 4;
  const int bh = blockIdx.y;
  const int m0 = blockIdx.x * 16;
  const int ds0 = wave * 128;

  const _Float16* qbase = Qh + ((size_t)bh * 1024 + m0) * 512;
  const _Float16* kbase = Kh + (size_t)bh * 1024 * 512;
  const _Float16* vbase = Vt + (size_t)bh * 512 * 1024;

  // Preload Q A-fragments over this wave's d-slice (reused all key blocks).
  v16h qf[4];
#pragma unroll
  for (int dt = 0; dt < 4; ++dt)
    qf[dt] = frag_f16(qbase + (size_t)lm * 512 + ds0 + dt * 32 + hs * 8);

  v8f oacc[8] = {};

  if (tid < 16) { rowmax[tid] = -1e30f; rowsum[tid] = 0.0f; }
  __syncthreads();

  for (int n0k = 0; n0k < 1024; n0k += 32) {
    // Non-blocking prefetch of next key block (global_prefetch_b8).
    if (n0k + 32 < 1024) {
      __builtin_prefetch(kbase + (size_t)(n0k + 32 + lm) * 512 + ds0, 0, 1);
      __builtin_prefetch(vbase + (size_t)(ds0 + lm) * 1024 + n0k + 32, 0, 1);
    }
    // ---- S partial = Q[:,dslice] . K[n0k:n0k+32, dslice]^T  (2 tiles) ----
    v8f s0 = {}, s1 = {};
#pragma unroll
    for (int dt = 0; dt < 4; ++dt) {
      v16h b0 = frag_f16(kbase + (size_t)(n0k + lm) * 512 + ds0 + dt * 32 + hs * 8);
      s0 = WMMA(qf[dt], b0, s0);
      v16h b1 = frag_f16(kbase + (size_t)(n0k + 16 + lm) * 512 + ds0 + dt * 32 + hs * 8);
      s1 = WMMA(qf[dt], b1, s1);
    }
    float* sp = Spart + wave * 512;
#pragma unroll
    for (int r = 0; r < 8; ++r) {
      const int m = r + hs * 8;
      sp[m * 32 + lm] = s0[r];
      sp[m * 32 + 16 + lm] = s1[r];
    }
    __syncthreads();

    // ---- online softmax: thread m owns row m, sums the 4 wave partials ----
    if (tid < 16) {
      const int m = tid;
      const float mold = rowmax[m], lold = rowsum[m];
      float sv[32];
      float smax = mold;
#pragma unroll
      for (int n = 0; n < 32; ++n) {
        float s = Spart[m * 32 + n] + Spart[512 + m * 32 + n] +
                  Spart[1024 + m * 32 + n] + Spart[1536 + m * 32 + n];
        sv[n] = s;
        smax = fmaxf(smax, s);
      }
      const float alpha = __expf(mold - smax);
      float lsum = lold * alpha;
#pragma unroll
      for (int n = 0; n < 32; ++n) {
        const float p = __expf(sv[n] - smax);
        lsum += p;
        Ptile[m * 32 + n] = (_Float16)p;
      }
      rowmax[m] = smax; rowsum[m] = lsum; alphaS[m] = alpha;
    }
    __syncthreads();

    // ---- rescale O accumulators, then O += P . V over this d-slice ----
#pragma unroll
    for (int r = 0; r < 8; ++r) {
      const float am = alphaS[r + hs * 8];
#pragma unroll
      for (int dt = 0; dt < 8; ++dt) oacc[dt][r] *= am;
    }
    v16h pf = frag_f16(Ptile + lm * 32 + hs * 8);  // P as A operand (16x32)
#pragma unroll
    for (int dt = 0; dt < 8; ++dt) {
      // B[k][n] = V[key][dcol] = Vt[dcol][key]: lane holds Vt row, contig in t
      v16h bf = frag_f16(vbase + (size_t)(ds0 + dt * 16 + lm) * 1024 + n0k + hs * 8);
      oacc[dt] = WMMA(pf, bf, oacc[dt]);
    }
    __syncthreads();
  }

  // ---- normalize and store to Oh (B,T,H*512) ----
  const int b = bh >> 3, h = bh & 7;
#pragma unroll
  for (int r = 0; r < 8; ++r) {
    const float inv = 1.0f / rowsum[r + hs * 8];
    const int t = m0 + r + hs * 8;
#pragma unroll
    for (int dt = 0; dt < 8; ++dt) {
      const int d = ds0 + dt * 16 + lm;
      Oh[((size_t)b * 1024 + t) * 4096 + h * 512 + d] = (_Float16)(oacc[dt][r] * inv);
    }
  }
}

// ---------------------------------------------------------------------------
// Kernel 3: out = Oh . Woh^T + bo (f32 out). grid = (8, 64); block = 128.
// Each wave: 16(M) x 64(N), K over 4096 in steps of 32 (128x4 WMMAs).
// ---------------------------------------------------------------------------
__global__ __launch_bounds__(128) void out_proj_kernel(
    const _Float16* __restrict__ Oh, const _Float16* __restrict__ Woh,
    const float* __restrict__ bo, float* __restrict__ out) {
  const int lane = threadIdx.x & 31, wave = threadIdx.x >> 5;
  const int lm = lane & 15, hs = lane >> 4;
  const int m0 = (blockIdx.y * 4 + wave) * 16;
  const int n0 = blockIdx.x * 64;

  v8f acc[4] = {};
  const _Float16* orow = Oh + (size_t)(m0 + lm) * 4096;
#pragma unroll 2
  for (int kk = 0; kk < 4096; kk += 32) {
    v16h a = frag_f16(orow + kk + hs * 8);
#pragma unroll
    for (int j = 0; j < 4; ++j) {
      v16h bfr = frag_f16(Woh + (size_t)(n0 + j * 16 + lm) * 4096 + kk + hs * 8);
      acc[j] = WMMA(a, bfr, acc[j]);
    }
  }
#pragma unroll
  for (int j = 0; j < 4; ++j) {
    const int n = n0 + j * 16 + lm;
    const float bias = bo[n];
#pragma unroll
    for (int r = 0; r < 8; ++r) {
      const int m = m0 + r + hs * 8;
      out[(size_t)m * 512 + n] = acc[j][r] + bias;
    }
  }
}

// ---------------------------------------------------------------------------
extern "C" void kernel_launch(void* const* d_in, const int* in_sizes, int n_in,
                              void* d_out, int out_size, void* d_ws, size_t ws_size,
                              hipStream_t stream) {
  (void)in_sizes; (void)n_in; (void)out_size; (void)ws_size;
  const float* x  = (const float*)d_in[0];
  const float* Wq = (const float*)d_in[1];
  const float* Wk = (const float*)d_in[2];
  const float* Wv = (const float*)d_in[3];
  const float* Wo = (const float*)d_in[4];
  const float* bo = (const float*)d_in[5];
  float* out = (float*)d_out;

  char* ws = (char*)d_ws;
  _Float16* Qh  = (_Float16*)(ws);
  _Float16* Kh  = (_Float16*)(ws + (size_t)33554432);
  _Float16* Vt  = (_Float16*)(ws + (size_t)67108864);
  _Float16* Oh  = (_Float16*)(ws + (size_t)100663296);
  _Float16* Xh  = (_Float16*)(ws + (size_t)134217728);
  _Float16* Wqh = (_Float16*)(ws + (size_t)138412032);
  _Float16* Wkh = (_Float16*)(ws + (size_t)142606336);
  _Float16* Wvh = (_Float16*)(ws + (size_t)146800640);
  _Float16* Woh = (_Float16*)(ws + (size_t)150994944);

  cvt_f16_kernel<<<dim3(2048, 5), 256, 0, stream>>>(x, Wq, Wk, Wv, Wo,
                                                    Xh, Wqh, Wkh, Wvh, Woh);
  qkv_proj_kernel<<<dim3(64, 64, 3), 128, 0, stream>>>(Xh, Wqh, Wkh, Wvh,
                                                       Qh, Kh, Vt);
  flash_attn_kernel<<<dim3(64, 32), 128, 0, stream>>>(Qh, Kh, Vt, Oh);
  out_proj_kernel<<<dim3(8, 64), 128, 0, stream>>>(Oh, Woh, bo, out);
}
